// ClaPointNet2SSG_61418032333276
// MI455X (gfx1250) — matrix operand order repaired and checked by reading, not compiled, not run on hardware
//
#include <hip/hip_runtime.h>
#include <hip/hip_bf16.h>

typedef __attribute__((ext_vector_type(16))) __bf16 v16bf;
typedef __attribute__((ext_vector_type(8)))  __bf16 v8bf;
typedef __attribute__((ext_vector_type(8)))  float  v8f;

#define EPS_BN 1e-5f

// ---------------------------------------------------------------------------
// x (B,3,N) -> xyz (B,N,3)
__global__ void transpose_kernel(const float* __restrict__ x, float* __restrict__ xyz,
                                 int B, int N) {
  int t = blockIdx.x * 256 + threadIdx.x;
  int total = B * N * 3;
  if (t >= total) return;
  int b = t / (N * 3);
  int r = t % (N * 3);
  int n = r / 3, c = r % 3;
  xyz[t] = x[((size_t)b * 3 + c) * N + n];
}

// ---------------------------------------------------------------------------
// Farthest point sampling: one block (256 thr) per batch, dist in LDS.
__global__ void fps_kernel(const float* __restrict__ xyz, int* __restrict__ fidx,
                           int N, int npoint) {
  __shared__ float dist[2048];
  __shared__ float rv[256];
  __shared__ int   ri[256];
  __shared__ int   s_far;
  const int b = blockIdx.x, tid = threadIdx.x;
  for (int n = tid; n < N; n += 256) dist[n] = 1e10f;
  if (tid == 0) s_far = 0;
  __syncthreads();
  for (int it = 0; it < npoint; ++it) {
    int far = s_far;
    if (tid == 0) fidx[b * npoint + it] = far;
    const float cx = xyz[((size_t)b * N + far) * 3 + 0];
    const float cy = xyz[((size_t)b * N + far) * 3 + 1];
    const float cz = xyz[((size_t)b * N + far) * 3 + 2];
    float bestv = -1.0f; int besti = 0;
    for (int n = tid; n < N; n += 256) {
      float dx = xyz[((size_t)b * N + n) * 3 + 0] - cx;
      float dy = xyz[((size_t)b * N + n) * 3 + 1] - cy;
      float dz = xyz[((size_t)b * N + n) * 3 + 2] - cz;
      float d = dx * dx + dy * dy + dz * dz;
      float m = fminf(dist[n], d);
      dist[n] = m;
      if (m > bestv) { bestv = m; besti = n; }
    }
    rv[tid] = bestv; ri[tid] = besti;
    __syncthreads();
    for (int s = 128; s > 0; s >>= 1) {
      if (tid < s) {
        if (rv[tid + s] > rv[tid] ||
            (rv[tid + s] == rv[tid] && ri[tid + s] < ri[tid])) {
          rv[tid] = rv[tid + s]; ri[tid] = ri[tid + s];
        }
      }
      __syncthreads();
    }
    if (tid == 0) s_far = ri[0];
    __syncthreads();
  }
}

// ---------------------------------------------------------------------------
// new_xyz (B,S,3) = xyz[fidx]
__global__ void gather_kernel(const float* __restrict__ xyz, const int* __restrict__ fidx,
                              float* __restrict__ out, int N, int S, int total) {
  int t = blockIdx.x * 256 + threadIdx.x;
  if (t >= total) return;
  int b = t / (S * 3);
  int r = t % (S * 3);
  int s = r / 3, c = r % 3;
  out[t] = xyz[((size_t)b * N + fidx[b * S + s]) * 3 + c];
}

// ---------------------------------------------------------------------------
// first-K in-radius indices (ascending), padded with first hit. One thread per (b,s).
__global__ void query_ball_kernel(const float* __restrict__ xyz, const float* __restrict__ nxyz,
                                  int* __restrict__ idx, int BS, int N, int S, int K, float r2) {
  int t = blockIdx.x * 256 + threadIdx.x;
  if (t >= BS) return;
  int b = t / S;
  const float cx = nxyz[(size_t)t * 3 + 0];
  const float cy = nxyz[(size_t)t * 3 + 1];
  const float cz = nxyz[(size_t)t * 3 + 2];
  int cnt = 0, first = 0;
  for (int n = 0; n < N && cnt < K; ++n) {
    float dx = xyz[((size_t)b * N + n) * 3 + 0] - cx;
    float dy = xyz[((size_t)b * N + n) * 3 + 1] - cy;
    float dz = xyz[((size_t)b * N + n) * 3 + 2] - cz;
    float d = dx * dx + dy * dy + dz * dz;
    if (d <= r2) {
      if (cnt == 0) first = n;
      idx[(size_t)t * K + cnt] = n;
      ++cnt;
    }
  }
  for (; cnt < K; ++cnt) idx[(size_t)t * K + cnt] = first;
}

// ---------------------------------------------------------------------------
// Build bf16 A-matrix rows = (b,s,k): [xyz[idx]-new_xyz | feat[idx]] zero-padded to Kpad.
__global__ void group_kernel(const float* __restrict__ xyz, const float* __restrict__ nxyz,
                             const int* __restrict__ idx, const float* __restrict__ feat,
                             __bf16* __restrict__ A, int Np, int S, int K, int Cf,
                             int Kpad, int total) {
  int t = blockIdx.x * 256 + threadIdx.x;
  if (t >= total) return;
  int col = t % Kpad;
  int row = t / Kpad;
  int s = (row / K) % S;
  int b = row / (K * S);
  int p = idx[row];
  float v = 0.0f;
  if (col < 3)
    v = xyz[((size_t)b * Np + p) * 3 + col] - nxyz[((size_t)b * S + s) * 3 + col];
  else if (col < 3 + Cf)
    v = feat[((size_t)b * Np + p) * Cf + (col - 3)];
  A[t] = (__bf16)v;
}

// group_all: rows = (b,j): [xyz | feat] (no centering), padded to Kpad.
__global__ void group_all_kernel(const float* __restrict__ xyz, const float* __restrict__ feat,
                                 __bf16* __restrict__ A, int Cf, int Kpad, int total) {
  int t = blockIdx.x * 256 + threadIdx.x;
  if (t >= total) return;
  int col = t % Kpad;
  int row = t / Kpad;
  float v = 0.0f;
  if (col < 3)           v = xyz[(size_t)row * 3 + col];
  else if (col < 3 + Cf) v = feat[(size_t)row * Cf + (col - 3)];
  A[t] = (__bf16)v;
}

// ---------------------------------------------------------------------------
__global__ void conv_w_kernel(const float* __restrict__ W, __bf16* __restrict__ Wb,
                              int O, int C, int Kpad) {
  int t = blockIdx.x * 256 + threadIdx.x;
  if (t >= O * Kpad) return;
  int o = t / Kpad, c = t % Kpad;
  Wb[t] = (c < C) ? (__bf16)W[(size_t)o * C + c] : (__bf16)0.0f;
}

__global__ void zero_f32_kernel(float* p, int n) {
  int t = blockIdx.x * 256 + threadIdx.x;
  if (t < n) p[t] = 0.0f;
}

__global__ void f32_to_bf16_kernel(const float* __restrict__ in, __bf16* __restrict__ out, int n) {
  int t = blockIdx.x * 256 + threadIdx.x;
  if (t < n) out[t] = (__bf16)in[t];
}

// ---------------------------------------------------------------------------
// Y[M,N](bf16) = A[M,Kpad] * W[N,Kpad]^T + bias, per-column sum/sumsq -> stats.
// Block = 4 waves; each wave computes a 16x64 output strip (4 accumulators),
// so the A fragment is loaded once and reused by 4 WMMAs per k-step.
__global__ void wmma_gemm_kernel(const __bf16* __restrict__ A, const __bf16* __restrict__ Wb,
                                 const float* __restrict__ bias, __bf16* __restrict__ Y,
                                 float* __restrict__ stats, int M, int N, int Kpad) {
  const int lane = threadIdx.x & 31;
  const int wave = threadIdx.x >> 5;
  const int half = lane >> 4;
  const int mr   = lane & 15;
  const int tileM  = (blockIdx.x * 4 + wave) * 16;
  const int tileN0 = blockIdx.y * 64;

  int arow  = tileM + mr;
  int arowc = (arow < M) ? arow : (M - 1);
  const __bf16* aptr = A  + (size_t)arowc * Kpad + half * 8;
  const __bf16* bptr = Wb + (size_t)(tileN0 + mr) * Kpad + half * 8;
  const size_t  bstp = (size_t)16 * Kpad;   // next 16 weight rows (N-subtile)

  v8f acc[4] = {v8f{}, v8f{}, v8f{}, v8f{}};
  for (int k = 0; k < Kpad; k += 32) {
    v8bf a0 = *(const v8bf*)(aptr + k);
    v8bf a1 = *(const v8bf*)(aptr + k + 16);
    v16bf av = __builtin_shufflevector(a0, a1, 0,1,2,3,4,5,6,7,8,9,10,11,12,13,14,15);
#pragma unroll
    for (int nt = 0; nt < 4; ++nt) {
      v8bf b0 = *(const v8bf*)(bptr + nt * bstp + k);
      v8bf b1 = *(const v8bf*)(bptr + nt * bstp + k + 16);
      v16bf bv = __builtin_shufflevector(b0, b1, 0,1,2,3,4,5,6,7,8,9,10,11,12,13,14,15);
      acc[nt] = __builtin_amdgcn_wmma_f32_16x16x32_bf16(false, av, false, bv,
                                                        (short)0, acc[nt], false, false);
    }
  }

  __shared__ float s_sum[64], s_sq[64];
  if (threadIdx.x < 64) { s_sum[threadIdx.x] = 0.0f; s_sq[threadIdx.x] = 0.0f; }
  __syncthreads();

#pragma unroll
  for (int nt = 0; nt < 4; ++nt) {
    const int col = tileN0 + nt * 16 + mr;
    const float bcol = bias[col];
    float ps = 0.0f, pq = 0.0f;
    for (int j = 0; j < 8; ++j) {
      int row = tileM + half * 8 + j;
      if (row < M) {
        float v = acc[nt][j] + bcol;
        Y[(size_t)row * N + col] = (__bf16)v;
        ps += v; pq += v * v;
      }
    }
    atomicAdd(&s_sum[nt * 16 + mr], ps);
    atomicAdd(&s_sq[nt * 16 + mr], pq);
  }
  __syncthreads();
  if (threadIdx.x < 64) {
    atomicAdd(&stats[tileN0 + threadIdx.x], s_sum[threadIdx.x]);
    atomicAdd(&stats[N + tileN0 + threadIdx.x], s_sq[threadIdx.x]);
  }
}

// ---------------------------------------------------------------------------
// BN (batch stats) + ReLU, write next layer's bf16 A (zero-padded to KpadNext).
__global__ void bn_relu_kernel(const __bf16* __restrict__ Y, const float* __restrict__ stats,
                               const float* __restrict__ gamma, const float* __restrict__ beta,
                               __bf16* __restrict__ Aout, int M, int N, int KpadNext,
                               float invCnt, int total) {
  int t = blockIdx.x * 256 + threadIdx.x;
  if (t >= total) return;
  int col = t % KpadNext;
  int row = t / KpadNext;
  float v = 0.0f;
  if (col < N) {
    float y   = (float)Y[(size_t)row * N + col];
    float m   = stats[col] * invCnt;
    float var = stats[N + col] * invCnt - m * m;
    v = gamma[col] * (y - m) * rsqrtf(var + EPS_BN) + beta[col];
    v = fmaxf(v, 0.0f);
  }
  Aout[t] = (__bf16)v;
}

// BN + ReLU + max over K -> f32 features (BS,N)
__global__ void bn_relu_maxpool_kernel(const __bf16* __restrict__ Y, const float* __restrict__ stats,
                                       const float* __restrict__ gamma, const float* __restrict__ beta,
                                       float* __restrict__ f, int BS, int K, int N, float invCnt) {
  int t = blockIdx.x * 256 + threadIdx.x;
  if (t >= BS * N) return;
  int bs = t / N, col = t % N;
  float m   = stats[col] * invCnt;
  float var = stats[N + col] * invCnt - m * m;
  float sc  = gamma[col] * rsqrtf(var + EPS_BN);
  float sh  = beta[col] - m * sc;
  float best = 0.0f;  // ReLU output >= 0
  for (int j = 0; j < K; ++j) {
    float y = (float)Y[((size_t)bs * K + j) * N + col];
    best = fmaxf(best, fmaxf(y * sc + sh, 0.0f));
  }
  f[t] = best;
}

// final 256->40 head (tiny; scalar)
__global__ void fc3_kernel(const __bf16* __restrict__ h, const float* __restrict__ W,
                           const float* __restrict__ b, float* __restrict__ out,
                           int M, int N, int K) {
  int t = blockIdx.x * 256 + threadIdx.x;
  if (t >= M * N) return;
  int i = t / N, o = t % N;
  float acc = b[o];
  for (int k = 0; k < K; ++k)
    acc += (float)h[(size_t)i * K + k] * W[(size_t)o * K + k];
  out[t] = acc;
}

// ---------------------------------------------------------------------------
extern "C" void kernel_launch(void* const* d_in, const int* in_sizes, int n_in,
                              void* d_out, int out_size, void* d_ws, size_t ws_size,
                              hipStream_t stream) {
  (void)in_sizes; (void)n_in; (void)out_size; (void)ws_size;
  const int B = 32;
  const float* X = (const float*)d_in[0];

  struct P { const float *W, *b, *g, *e; int O, C; };
  int pi = 1;
  auto nextP = [&](int O, int C) {
    P p; p.W = (const float*)d_in[pi]; p.b = (const float*)d_in[pi + 1];
    p.g = (const float*)d_in[pi + 2]; p.e = (const float*)d_in[pi + 3];
    pi += 4; p.O = O; p.C = C; return p;
  };
  P sa1_0 = nextP(64, 3),    sa1_1 = nextP(64, 64),   sa1_2 = nextP(128, 64);
  P sa2_0 = nextP(128, 131), sa2_1 = nextP(128, 128), sa2_2 = nextP(256, 128);
  P sa3_0 = nextP(256, 259), sa3_1 = nextP(512, 256), sa3_2 = nextP(1024, 512);
  P fc1   = nextP(512, 1024), fc2  = nextP(256, 512);
  const float* W3 = (const float*)d_in[pi];
  const float* b3 = (const float*)d_in[pi + 1];

  // bump allocator over d_ws
  char* ws = (char*)d_ws;
  size_t off = 0;
  auto alloc = [&](size_t bytes) -> void* {
    void* p = ws + off;
    off += (bytes + 255) & ~(size_t)255;
    return p;
  };
  const int M1 = B * 512 * 32, M2 = B * 128 * 64, M3 = B * 128;

  float*  xyz   = (float*) alloc((size_t)B * 2048 * 3 * 4);
  int*    fidx  = (int*)   alloc((size_t)B * 512 * 4);
  float*  nxyz1 = (float*) alloc((size_t)B * 512 * 3 * 4);
  int*    idx1  = (int*)   alloc((size_t)B * 512 * 32 * 4);
  float*  f1    = (float*) alloc((size_t)B * 512 * 128 * 4);
  float*  nxyz2 = (float*) alloc((size_t)B * 128 * 3 * 4);
  int*    idx2  = (int*)   alloc((size_t)B * 128 * 64 * 4);
  float*  f2    = (float*) alloc((size_t)B * 128 * 256 * 4);
  float*  f3    = (float*) alloc((size_t)B * 1024 * 4);
  float*  stats = (float*) alloc(2 * 1024 * 4);
  __bf16* Wb    = (__bf16*)alloc((size_t)1024 * 1024 * 2);
  __bf16* Abuf  = (__bf16*)alloc((size_t)M2 * 160 * 2);   // largest A: SA2 input
  __bf16* Ybuf  = (__bf16*)alloc((size_t)M1 * 128 * 2);   // largest Y: SA1/SA2 L2

#define CD(n) (((n) + 255) / 256)
  auto mlp = [&](const P& p, int M, int Kpad) {
    conv_w_kernel<<<CD(p.O * Kpad), 256, 0, stream>>>(p.W, Wb, p.O, p.C, Kpad);
    zero_f32_kernel<<<CD(2 * p.O), 256, 0, stream>>>(stats, 2 * p.O);
    dim3 g((unsigned)((M + 63) / 64), (unsigned)(p.O / 64));
    wmma_gemm_kernel<<<g, 128, 0, stream>>>(Abuf, Wb, p.b, Ybuf, stats, M, p.O, Kpad);
  };
  auto bnrelu = [&](const P& p, int M, int KpadNext, float invCnt) {
    int tt = M * KpadNext;
    bn_relu_kernel<<<CD(tt), 256, 0, stream>>>(Ybuf, stats, p.g, p.e, Abuf,
                                               M, p.O, KpadNext, invCnt, tt);
  };
  auto bnpool = [&](const P& p, int BS, int K, float* fout, float invCnt) {
    bn_relu_maxpool_kernel<<<CD(BS * p.O), 256, 0, stream>>>(Ybuf, stats, p.g, p.e,
                                                             fout, BS, K, p.O, invCnt);
  };

  // -------- input transpose --------
  transpose_kernel<<<CD(B * 2048 * 3), 256, 0, stream>>>(X, xyz, B, 2048);

  // -------- SA1: npoint=512, r=0.2, k=32, MLP 3->64->64->128 --------
  fps_kernel<<<B, 256, 0, stream>>>(xyz, fidx, 2048, 512);
  gather_kernel<<<CD(B * 512 * 3), 256, 0, stream>>>(xyz, fidx, nxyz1, 2048, 512, B * 512 * 3);
  query_ball_kernel<<<CD(B * 512), 256, 0, stream>>>(xyz, nxyz1, idx1, B * 512, 2048, 512, 32, 0.04f);
  group_kernel<<<CD(M1 * 32), 256, 0, stream>>>(xyz, nxyz1, idx1, nullptr, Abuf,
                                                2048, 512, 32, 0, 32, M1 * 32);
  const float inv1 = 1.0f / (float)M1;
  mlp(sa1_0, M1, 32);  bnrelu(sa1_0, M1, 64, inv1);
  mlp(sa1_1, M1, 64);  bnrelu(sa1_1, M1, 64, inv1);
  mlp(sa1_2, M1, 64);  bnpool(sa1_2, B * 512, 32, f1, inv1);

  // -------- SA2: npoint=128, r=0.4, k=64, MLP 131->128->128->256 --------
  fps_kernel<<<B, 256, 0, stream>>>(nxyz1, fidx, 512, 128);
  gather_kernel<<<CD(B * 128 * 3), 256, 0, stream>>>(nxyz1, fidx, nxyz2, 512, 128, B * 128 * 3);
  query_ball_kernel<<<CD(B * 128), 256, 0, stream>>>(nxyz1, nxyz2, idx2, B * 128, 512, 128, 64, 0.16f);
  group_kernel<<<CD(M2 * 160), 256, 0, stream>>>(nxyz1, nxyz2, idx2, f1, Abuf,
                                                 512, 128, 64, 128, 160, M2 * 160);
  const float inv2 = 1.0f / (float)M2;
  mlp(sa2_0, M2, 160); bnrelu(sa2_0, M2, 128, inv2);
  mlp(sa2_1, M2, 128); bnrelu(sa2_1, M2, 128, inv2);
  mlp(sa2_2, M2, 128); bnpool(sa2_2, B * 128, 64, f2, inv2);

  // -------- SA3: group_all, MLP 259->256->512->1024 --------
  group_all_kernel<<<CD(M3 * 288), 256, 0, stream>>>(nxyz2, f2, Abuf, 256, 288, M3 * 288);
  const float inv3 = 1.0f / (float)M3;
  mlp(sa3_0, M3, 288); bnrelu(sa3_0, M3, 256, inv3);
  mlp(sa3_1, M3, 256); bnrelu(sa3_1, M3, 512, inv3);
  mlp(sa3_2, M3, 512); bnpool(sa3_2, B, 128, f3, inv3);

  // -------- FC head --------
  f32_to_bf16_kernel<<<CD(B * 1024), 256, 0, stream>>>(f3, Abuf, B * 1024);
  const float invB = 1.0f / (float)B;
  mlp(fc1, B, 1024); bnrelu(fc1, B, 512, invB);
  mlp(fc2, B, 512);  bnrelu(fc2, B, 256, invB);
  fc3_kernel<<<CD(B * 40), 256, 0, stream>>>(Abuf, W3, b3, (float*)d_out, B, 40, 256);
#undef CD
}
